// Encoder_block_11553462026609
// MI455X (gfx1250) — matrix operand order repaired and checked by reading, not compiled, not run on hardware
//
#include <hip/hip_runtime.h>
#include <hip/hip_bf16.h>

// ---------------------------------------------------------------------------
// Types & helpers
// ---------------------------------------------------------------------------
typedef __attribute__((ext_vector_type(16))) __bf16 bf16x16;
typedef __attribute__((ext_vector_type(8)))  float  floatx8;
typedef __attribute__((__vector_size__(16))) int    i32x4;

union Frag {
  uint4          u[2];   // 32 bytes = 16 bf16
  bf16x16        v;
  unsigned short h[16];
};

__device__ __forceinline__ unsigned short f2bf(float f) {
  unsigned int u = __float_as_uint(f);
  u += 0x7FFFu + ((u >> 16) & 1u);          // round-to-nearest-even
  return (unsigned short)(u >> 16);
}
__device__ __forceinline__ float bf2f(unsigned short h) {
  return __uint_as_float(((unsigned int)h) << 16);
}
__device__ __forceinline__ floatx8 zero8() {
  floatx8 z = {0.f, 0.f, 0.f, 0.f, 0.f, 0.f, 0.f, 0.f};
  return z;
}
__device__ __forceinline__ floatx8 wmma_bf16(const Frag& a, const Frag& b, floatx8 c) {
  return __builtin_amdgcn_wmma_f32_16x16x32_bf16(
      /*neg_a=*/false, a.v, /*neg_b=*/false, b.v,
      /*c_mod=*/(short)0, c, /*reuse_a=*/false, /*reuse_b=*/false);
}

// Model constants (fixed by reference)
#define EMB   1024
#define NHEAD 16
#define HEAD  64
#define DFF   4096
#define SEQB  8192        // B*T = 4*2048
#define TLEN  2048

// ---------------------------------------------------------------------------
// Async global->LDS copy (CDNA5). Falls back to plain copy if builtins absent.
// ---------------------------------------------------------------------------
#define AS1 __attribute__((address_space(1)))
#define AS3 __attribute__((address_space(3)))

#if defined(__has_builtin)
#if __has_builtin(__builtin_amdgcn_global_load_async_to_lds_b128) && \
    __has_builtin(__builtin_amdgcn_s_wait_asynccnt)
#define USE_ASYNC 1
#endif
#endif
#ifndef USE_ASYNC
#define USE_ASYNC 0
#endif

__device__ __forceinline__ void cp16(const unsigned short* g, unsigned short* l) {
#if USE_ASYNC
  __builtin_amdgcn_global_load_async_to_lds_b128(
      (AS1 i32x4*)g, (AS3 i32x4*)l, 0, 0);
#else
  *(uint4*)l = *(const uint4*)g;
#endif
}
__device__ __forceinline__ void wait_async_keep8() {
#if USE_ASYNC
  __builtin_amdgcn_s_wait_asynccnt(8);
#endif
}
__device__ __forceinline__ void wait_async_all() {
#if USE_ASYNC
  __builtin_amdgcn_s_wait_asynccnt(0);
#endif
}

// ---------------------------------------------------------------------------
// Pack kernels: fp32 -> bf16, weights stored TRANSPOSED [N][K]
// ---------------------------------------------------------------------------
__global__ void pack_f32_to_bf16(const float* __restrict__ in,
                                 unsigned short* __restrict__ out, int n) {
  int i = blockIdx.x * 256 + threadIdx.x;
  if (i < n) out[i] = f2bf(in[i]);
}

// generic W [K,N] fp32 -> WT bf16 [N,K]
__global__ void pack_w_t(const float* __restrict__ W, unsigned short* __restrict__ out,
                         int K, int N) {
  int i = blockIdx.x * 256 + threadIdx.x;       // over N*K
  int n = i / K, k = i % K;
  out[i] = f2bf(W[(size_t)k * N + n]);
}

// Wq/Wk/Wv: (H, EMB, HEAD) each -> WqkvT bf16 [3*EMB][EMB]; row = sec*1024+h*64+kk
__global__ void pack_qkv_w_t(const float* __restrict__ Wq, const float* __restrict__ Wk,
                             const float* __restrict__ Wv, unsigned short* __restrict__ out) {
  int i = blockIdx.x * 256 + threadIdx.x;       // over 3*EMB*EMB
  int col = i / EMB, d = i % EMB;               // col in [0, 3*EMB)
  int sec = col / EMB;                          // 0..2
  int w = col % EMB;
  int h = w / HEAD, kk = w % HEAD;
  const float* W = (sec == 0) ? Wq : (sec == 1) ? Wk : Wv;
  out[i] = f2bf(W[((size_t)h * EMB + d) * HEAD + kk]);
}

// ---------------------------------------------------------------------------
// Tiled bf16 WMMA GEMM: C[M,N] = A[M,K] * BT[N,K]^T (+bias)(+gelu)
//   128 threads (4 waves), block tile 128x128, wave tile 64x64, K-step 32.
//   Double-buffered LDS, async global->LDS staging.
//   EPI: 0 = store bf16   1 = +bias, store fp32   2 = +bias, GELU, store bf16
// ---------------------------------------------------------------------------
#define BM  128
#define BN  128
#define BK  32
#define LDK 40   // BK + 8 pad (shorts) ; row stride 80B (16B aligned)

template <int EPI>
__global__ void __launch_bounds__(128, 1)
gemm_bf16(const unsigned short* __restrict__ A, const unsigned short* __restrict__ BT,
          const float* __restrict__ bias, void* __restrict__ Cout,
          int M, int N, int K) {
  __shared__ __align__(16) unsigned short As[2][BM][LDK];
  __shared__ __align__(16) unsigned short Bs[2][BN][LDK];

  const int tid  = threadIdx.x;                 // 0..127
  const int lane = tid & 31;
  const int wave = tid >> 5;                    // 0..3
  const int bm = blockIdx.y * BM;
  const int bn = blockIdx.x * BN;
  const int wm = (wave & 1) * 64;
  const int wn = (wave >> 1) * 64;

  const unsigned short* Ag = A  + (size_t)(bm + tid) * K;   // this thread's A row
  const unsigned short* Bg = BT + (size_t)(bn + tid) * K;   // this thread's B row

  floatx8 acc[4][4];
#pragma unroll
  for (int i = 0; i < 4; ++i)
#pragma unroll
    for (int j = 0; j < 4; ++j) acc[i][j] = zero8();

  const int m    = lane & 15;
  const int aoff = (lane < 16) ? 0 : 16;
  const int boff = (lane < 16) ? 0 : 16;
  const int nk   = K / BK;

  // prologue: stage k-step 0 into buffer 0
#pragma unroll
  for (int i = 0; i < 4; ++i) cp16(Ag + 8 * i, &As[0][tid][8 * i]);
#pragma unroll
  for (int i = 0; i < 4; ++i) cp16(Bg + 8 * i, &Bs[0][tid][8 * i]);

  for (int ik = 0; ik < nk; ++ik) {
    const int cur = ik & 1;
    if (ik + 1 < nk) {
      const int nxt = cur ^ 1, k1 = (ik + 1) * BK;
#pragma unroll
      for (int i = 0; i < 4; ++i) cp16(Ag + k1 + 8 * i, &As[nxt][tid][8 * i]);
#pragma unroll
      for (int i = 0; i < 4; ++i) cp16(Bg + k1 + 8 * i, &Bs[nxt][tid][8 * i]);
      wait_async_keep8();          // oldest 8 (buffer cur) complete, newest 8 in flight
    } else {
      wait_async_all();
    }
    __syncthreads();               // all waves' staging of buffer cur visible

    Frag a[4], b[4];
#pragma unroll
    for (int i = 0; i < 4; ++i) {
      const unsigned short* ap = &As[cur][wm + i * 16 + m][0];
      a[i].u[0] = *(const uint4*)(ap + aoff);        // K = aoff..aoff+7
      a[i].u[1] = *(const uint4*)(ap + aoff + 8);    // K = aoff+8..aoff+15
      const unsigned short* bp = &Bs[cur][wn + i * 16 + m][0];
      b[i].u[0] = *(const uint4*)(bp + boff);        // K = boff..boff+7
      b[i].u[1] = *(const uint4*)(bp + boff + 8);    // K = boff+8..boff+15
    }
#pragma unroll
    for (int i = 0; i < 4; ++i)
#pragma unroll
      for (int j = 0; j < 4; ++j)
        acc[i][j] = wmma_bf16(a[i], b[j], acc[i][j]);
    __syncthreads();               // everyone done reading buffer cur
  }

  // ---- epilogue (C layout: VGPR r -> row r (lanes 0-15) / r+8 (lanes 16-31)) ----
#pragma unroll
  for (int i = 0; i < 4; ++i)
#pragma unroll
    for (int j = 0; j < 4; ++j) {
      int row0 = bm + wm + i * 16 + ((lane < 16) ? 0 : 8);
      int col  = bn + wn + j * 16 + (lane & 15);
#pragma unroll
      for (int r = 0; r < 8; ++r) {
        float v = acc[i][j][r];
        size_t idx = (size_t)(row0 + r) * N + col;
        if (EPI == 0) {
          ((unsigned short*)Cout)[idx] = f2bf(v);
        } else if (EPI == 1) {
          ((float*)Cout)[idx] = v + bias[col];
        } else {
          float hgl = v + bias[col];
          hgl = 0.5f * hgl * (1.0f + erff(hgl * 0.70710678118654752f));
          ((unsigned short*)Cout)[idx] = f2bf(hgl);
        }
      }
    }
}

// ---------------------------------------------------------------------------
// Flash attention: one (b, h, 64-query tile) per 128-thread block (4 waves).
// QKV bf16 [SEQB, 3*EMB] (cols: q | k | v, each h*64+kk). Out bf16 [SEQB, EMB].
// ---------------------------------------------------------------------------
__global__ void __launch_bounds__(128)
attn_flash(const unsigned short* __restrict__ QKV, unsigned short* __restrict__ Out) {
  __shared__ __align__(16) unsigned short Ks[64][72];      // [s][kk]
  __shared__ __align__(16) unsigned short Vt[64][72];      // [kk][s]  (transposed)
  __shared__ __align__(16) unsigned short Ps[4][16][72];   // per-wave P staging [m][s]

  const int tid  = threadIdx.x;
  const int lane = tid & 31;
  const int wave = tid >> 5;
  const int qt   = blockIdx.x & 31;          // T/64 tiles
  const int bh   = blockIdx.x >> 5;
  const int b    = bh >> 4;
  const int h    = bh & 15;
  const size_t base = (size_t)b * TLEN * (3 * EMB);
  const int qbase = qt * 64 + wave * 16;

  const int m    = lane & 15;
  const int aoff = (lane < 16) ? 0 : 16;
  const int boff = (lane < 16) ? 0 : 16;

  // Q fragments (16x64 as two 16x32 A-frags), pre-scaled by 1/sqrt(HEAD)=0.125
  Frag aq[2];
  {
    const unsigned short* qrow = QKV + base + (size_t)(qbase + m) * (3 * EMB) + h * HEAD;
#pragma unroll
    for (int s = 0; s < 2; ++s) {
      aq[s].u[0] = *(const uint4*)(qrow + s * 32 + aoff);
      aq[s].u[1] = *(const uint4*)(qrow + s * 32 + aoff + 8);
#pragma unroll
      for (int i = 0; i < 16; ++i) aq[s].h[i] = f2bf(bf2f(aq[s].h[i]) * 0.125f);
    }
  }

  float mrun[8], lrun[8];
#pragma unroll
  for (int r = 0; r < 8; ++r) { mrun[r] = -1e30f; lrun[r] = 0.f; }
  floatx8 o[4];
#pragma unroll
  for (int ot = 0; ot < 4; ++ot) o[ot] = zero8();

  for (int s0 = 0; s0 < TLEN; s0 += 64) {
    // ---- cooperative load of K (row-major) and V (transposed) tiles ----
    {
      int row = tid >> 1, half = (tid & 1) * 32;
      const unsigned short* kp = QKV + base + (size_t)(s0 + row) * (3 * EMB) + EMB + h * HEAD + half;
      *(uint4*)&Ks[row][half]     = ((const uint4*)kp)[0];
      *(uint4*)&Ks[row][half + 8] = ((const uint4*)kp)[1];
      const unsigned short* vp = QKV + base + (size_t)(s0 + row) * (3 * EMB) + 2 * EMB + h * HEAD + half;
      uint4 d0 = ((const uint4*)vp)[0];
      uint4 d1 = ((const uint4*)vp)[1];
      const unsigned short* e0 = (const unsigned short*)&d0;
      const unsigned short* e1 = (const unsigned short*)&d1;
#pragma unroll
      for (int i = 0; i < 8; ++i) Vt[half + i][row] = e0[i];
#pragma unroll
      for (int i = 0; i < 8; ++i) Vt[half + 8 + i][row] = e1[i];
    }
    __syncthreads();

    // ---- scores 16x64 = Q (16x64) x K^T ----
    floatx8 sc[4];
#pragma unroll
    for (int st = 0; st < 4; ++st) {
      const unsigned short* kp = &Ks[st * 16 + m][0];
      Frag bk0, bk1;
      bk0.u[0] = *(const uint4*)(kp + boff);
      bk0.u[1] = *(const uint4*)(kp + boff + 8);
      bk1.u[0] = *(const uint4*)(kp + 32 + boff);
      bk1.u[1] = *(const uint4*)(kp + 32 + boff + 8);
      floatx8 c = zero8();
      c = wmma_bf16(aq[0], bk0, c);
      c = wmma_bf16(aq[1], bk1, c);
      sc[st] = c;
    }

    // ---- online softmax (rows live in 16-lane N-groups) ----
    float mnew[8], scale[8];
#pragma unroll
    for (int r = 0; r < 8; ++r) {
      float tm = fmaxf(fmaxf(sc[0][r], sc[1][r]), fmaxf(sc[2][r], sc[3][r]));
#pragma unroll
      for (int msk = 8; msk >= 1; msk >>= 1) tm = fmaxf(tm, __shfl_xor(tm, msk, 32));
      mnew[r]  = fmaxf(mrun[r], tm);
      scale[r] = __expf(mrun[r] - mnew[r]);
      mrun[r]  = mnew[r];
    }
#pragma unroll
    for (int st = 0; st < 4; ++st)
#pragma unroll
      for (int r = 0; r < 8; ++r) sc[st][r] = __expf(sc[st][r] - mnew[r]);
#pragma unroll
    for (int r = 0; r < 8; ++r) {
      float rs = sc[0][r] + sc[1][r] + sc[2][r] + sc[3][r];
#pragma unroll
      for (int msk = 8; msk >= 1; msk >>= 1) rs += __shfl_xor(rs, msk, 32);
      lrun[r] = lrun[r] * scale[r] + rs;
    }
#pragma unroll
    for (int ot = 0; ot < 4; ++ot)
#pragma unroll
      for (int r = 0; r < 8; ++r) o[ot][r] *= scale[r];

    // ---- P: C-layout -> LDS -> A-layout ----
    {
      int prow = (lane < 16) ? 0 : 8;
#pragma unroll
      for (int st = 0; st < 4; ++st)
#pragma unroll
        for (int r = 0; r < 8; ++r)
          Ps[wave][prow + r][st * 16 + m] = f2bf(sc[st][r]);
    }
    Frag ap[2];
    {
      const unsigned short* pp = &Ps[wave][m][0];
      ap[0].u[0] = *(const uint4*)(pp + aoff);
      ap[0].u[1] = *(const uint4*)(pp + aoff + 8);
      ap[1].u[0] = *(const uint4*)(pp + 32 + aoff);
      ap[1].u[1] = *(const uint4*)(pp + 32 + aoff + 8);
    }

    // ---- O += P (16x64) x V (64x64) ----
#pragma unroll
    for (int ot = 0; ot < 4; ++ot) {
      const unsigned short* vp = &Vt[ot * 16 + m][0];
      Frag bv0, bv1;
      bv0.u[0] = *(const uint4*)(vp + boff);
      bv0.u[1] = *(const uint4*)(vp + boff + 8);
      bv1.u[0] = *(const uint4*)(vp + 32 + boff);
      bv1.u[1] = *(const uint4*)(vp + 32 + boff + 8);
      o[ot] = wmma_bf16(ap[0], bv0, o[ot]);
      o[ot] = wmma_bf16(ap[1], bv1, o[ot]);
    }
    __syncthreads();
  }

  // ---- finalize: O /= l, store heads concat [SEQB, EMB] bf16 ----
#pragma unroll
  for (int ot = 0; ot < 4; ++ot) {
    int col = h * HEAD + ot * 16 + (lane & 15);
    int r0  = (lane < 16) ? 0 : 8;
#pragma unroll
    for (int r = 0; r < 8; ++r) {
      int row = qbase + r0 + r;
      float val = o[ot][r] / lrun[r];
      Out[((size_t)b * TLEN + row) * EMB + col] = f2bf(val);
    }
  }
}

// ---------------------------------------------------------------------------
// out = xres + LayerNorm(y)*g + beta  (row = 1024). Optionally emit bf16 copy.
// ---------------------------------------------------------------------------
__global__ void __launch_bounds__(256)
ln_residual(const float* __restrict__ xres, const float* __restrict__ y,
            const float* __restrict__ g, const float* __restrict__ beta,
            float* __restrict__ outf, unsigned short* __restrict__ outbf) {
  __shared__ float red[256];
  const int row = blockIdx.x, tid = threadIdx.x;
  const float* yr = y + (size_t)row * EMB;

  float v[4], s = 0.f;
#pragma unroll
  for (int i = 0; i < 4; ++i) { v[i] = yr[tid + i * 256]; s += v[i]; }
  red[tid] = s;
  __syncthreads();
  for (int off = 128; off > 0; off >>= 1) {
    if (tid < off) red[tid] += red[tid + off];
    __syncthreads();
  }
  const float mu = red[0] * (1.f / EMB);
  __syncthreads();

  float s2 = 0.f;
#pragma unroll
  for (int i = 0; i < 4; ++i) { float d = v[i] - mu; s2 += d * d; }
  red[tid] = s2;
  __syncthreads();
  for (int off = 128; off > 0; off >>= 1) {
    if (tid < off) red[tid] += red[tid + off];
    __syncthreads();
  }
  const float rstd = rsqrtf(red[0] * (1.f / EMB) + 1e-5f);

#pragma unroll
  for (int i = 0; i < 4; ++i) {
    int c = tid + i * 256;
    size_t idx = (size_t)row * EMB + c;
    float o = xres[idx] + (v[i] - mu) * rstd * g[c] + beta[c];
    if (outf)  outf[idx]  = o;
    if (outbf) outbf[idx] = f2bf(o);
  }
}

// ---------------------------------------------------------------------------
// Host launcher
// ---------------------------------------------------------------------------
extern "C" void kernel_launch(void* const* d_in, const int* in_sizes, int n_in,
                              void* d_out, int out_size, void* d_ws, size_t ws_size,
                              hipStream_t stream) {
  const float* x     = (const float*)d_in[0];
  const float* Wq    = (const float*)d_in[1];
  const float* Wk    = (const float*)d_in[2];
  const float* Wv    = (const float*)d_in[3];
  const float* Wo    = (const float*)d_in[4];
  const float* bo    = (const float*)d_in[5];
  const float* ln1_g = (const float*)d_in[6];
  const float* ln1_b = (const float*)d_in[7];
  const float* W1    = (const float*)d_in[8];
  const float* b1    = (const float*)d_in[9];
  const float* W2    = (const float*)d_in[10];
  const float* b2    = (const float*)d_in[11];
  const float* ln2_g = (const float*)d_in[12];
  const float* ln2_b = (const float*)d_in[13];

  char* ws = (char*)d_ws;
  const size_t MB = 1024ull * 1024ull;
  // Region A (64MB): Xb (16MB) + QKV (48MB); reused later by Hff (64MB).
  unsigned short* Xb    = (unsigned short*)(ws + 0);
  unsigned short* QKV   = (unsigned short*)(ws + 16 * MB);
  unsigned short* Hff   = (unsigned short*)(ws + 0);
  unsigned short* attn  = (unsigned short*)(ws + 64 * MB);   // 16MB
  float*          mha   = (float*)(ws + 80 * MB);            // 32MB, reused by ff
  float*          ff    = (float*)(ws + 80 * MB);
  float*          x1f   = (float*)(ws + 112 * MB);           // 32MB
  unsigned short* x1b   = (unsigned short*)(ws + 144 * MB);  // 16MB
  unsigned short* WqkvT = (unsigned short*)(ws + 160 * MB);  // 6MB  [3072][1024]
  unsigned short* WoT   = (unsigned short*)(ws + 166 * MB);  // 2MB  [1024][1024]
  unsigned short* W1T   = (unsigned short*)(ws + 168 * MB);  // 8MB  [4096][1024]
  unsigned short* W2T   = (unsigned short*)(ws + 176 * MB);  // 8MB  [1024][4096]

  // 1) packs (weights transposed to [N][K] bf16)
  pack_f32_to_bf16<<<(SEQB * EMB) / 256, 256, 0, stream>>>(x, Xb, SEQB * EMB);
  pack_qkv_w_t<<<(3 * EMB * EMB) / 256, 256, 0, stream>>>(Wq, Wk, Wv, WqkvT);
  pack_w_t<<<(EMB * EMB) / 256, 256, 0, stream>>>(Wo, WoT, EMB, EMB);
  pack_w_t<<<(EMB * DFF) / 256, 256, 0, stream>>>(W1, W1T, EMB, DFF);
  pack_w_t<<<(DFF * EMB) / 256, 256, 0, stream>>>(W2, W2T, DFF, EMB);

  // 2) QKV projection: [8192,1024] x [1024,3072] -> bf16
  gemm_bf16<0><<<dim3((3 * EMB) / BN, SEQB / BM), 128, 0, stream>>>(
      Xb, WqkvT, nullptr, QKV, SEQB, 3 * EMB, EMB);

  // 3) attention -> concat heads bf16 [8192,1024]
  attn_flash<<<4 * NHEAD * (TLEN / 64), 128, 0, stream>>>(QKV, attn);

  // 4) output projection + bias -> fp32 mha
  gemm_bf16<1><<<dim3(EMB / BN, SEQB / BM), 128, 0, stream>>>(
      attn, WoT, bo, mha, SEQB, EMB, EMB);

  // 5) x1 = x + LN(mha)  (fp32 + bf16 copies)
  ln_residual<<<SEQB, 256, 0, stream>>>(x, mha, ln1_g, ln1_b, x1f, x1b);

  // 6) FFN up: +b1, GELU -> bf16 H [8192,4096]   (Hff overlaps dead Xb/QKV)
  gemm_bf16<2><<<dim3(DFF / BN, SEQB / BM), 128, 0, stream>>>(
      x1b, W1T, b1, Hff, SEQB, DFF, EMB);

  // 7) FFN down: +b2 -> fp32 ff
  gemm_bf16<1><<<dim3(EMB / BN, SEQB / BM), 128, 0, stream>>>(
      Hff, W2T, b2, ff, SEQB, EMB, DFF);

  // 8) out = x1 + LN(ff)
  ln_residual<<<SEQB, 256, 0, stream>>>(x1f, ff, ln2_g, ln2_b, (float*)d_out, nullptr);
}
